// AttentionFCHead_90615220011047
// MI455X (gfx1250) — matrix operand order repaired and checked by reading, not compiled, not run on hardware
//
#include <hip/hip_runtime.h>
#include <hip/hip_bf16.h>

#define N_ROI 512
#define M_REF 4096
#define FEATD 1024
#define NGRP  16
#define DGRP  64

typedef __attribute__((ext_vector_type(16))) __bf16       v16bf;
typedef __attribute__((ext_vector_type(8)))  float        v8f;
typedef __attribute__((ext_vector_type(4)))  float        v4f;
typedef __attribute__((ext_vector_type(4)))  unsigned int v4u;

union Frag {
    v16bf  v;
    v4u    u[2];
    __bf16 e[16];
};

static __device__ __forceinline__ v8f wmma_bf16(v16bf a, v16bf b, v8f c) {
    // D(f32 16x16) = A(bf16 16x32) * B(bf16 32x16) + C
    return __builtin_amdgcn_wmma_f32_16x16x32_bf16(false, a, false, b,
                                                   (short)0, c, false, false);
}

// ---------------------------------------------------------------------------
// out[i,j] = sum_f A[i,f] * W[j,f] (+ bias[j]), stored as bf16.
// transposed==0: out[i*ldOut + j]   (row-major [rows x 1024])
// transposed==1: out[j*ldOut + i]   (VT layout [1024 x rows])
// Block: 256 thr = 8 waves, each wave owns a 16x16 C tile; block tile 32x64.
// ---------------------------------------------------------------------------
__global__ __launch_bounds__(256)
void gemm_nt_bf16(const float* __restrict__ A, const float* __restrict__ W,
                  const float* __restrict__ bias, __bf16* __restrict__ out,
                  int ldOut, int transposed) {
    const int lane = threadIdx.x & 31;
    const int wave = threadIdx.x >> 5;
    const int l15  = lane & 15;
    const int hl   = lane >> 4;           // 16-lane half select
    const int row0 = blockIdx.x * 32 + (wave >> 2) * 16;
    const int col0 = blockIdx.y * 64 + (wave & 3) * 16;

    const float* Ar = A + (size_t)(row0 + l15) * FEATD;   // A row for this lane
    const float* Wr = W + (size_t)(col0 + l15) * FEATD;   // W row (= B column)

    v8f acc = {0.f, 0.f, 0.f, 0.f, 0.f, 0.f, 0.f, 0.f};

    for (int k = 0; k < FEATD; k += 32) {
        Frag fa, fb;
        // A fragment: lane<16 -> K k..k+7 & k+16..k+23 ; lane>=16 -> +8
        {
            const int ko = k + hl * 8;
            v4f a0 = *(const v4f*)(Ar + ko);
            v4f a1 = *(const v4f*)(Ar + ko + 4);
            v4f a2 = *(const v4f*)(Ar + ko + 16);
            v4f a3 = *(const v4f*)(Ar + ko + 20);
#pragma unroll
            for (int i = 0; i < 4; ++i) {
                fa.e[i]      = (__bf16)a0[i];
                fa.e[4 + i]  = (__bf16)a1[i];
                fa.e[8 + i]  = (__bf16)a2[i];
                fa.e[12 + i] = (__bf16)a3[i];
            }
        }
        // B fragment: lane<16 -> K k..k+7 & k+16..k+23 ; lane>=16 -> +8
        {
            const int ko = k + hl * 8;
            v4f b0 = *(const v4f*)(Wr + ko);
            v4f b1 = *(const v4f*)(Wr + ko + 4);
            v4f b2 = *(const v4f*)(Wr + ko + 16);
            v4f b3 = *(const v4f*)(Wr + ko + 20);
#pragma unroll
            for (int i = 0; i < 4; ++i) {
                fb.e[i]      = (__bf16)b0[i];
                fb.e[4 + i]  = (__bf16)b1[i];
                fb.e[8 + i]  = (__bf16)b2[i];
                fb.e[12 + i] = (__bf16)b3[i];
            }
        }
        if (k + 32 < FEATD) {
            __builtin_prefetch(Ar + k + 32, 0, 3);  // global_prefetch_b8
            __builtin_prefetch(Wr + k + 32, 0, 3);
        }
        acc = wmma_bf16(fa.v, fb.v, acc);
    }

    const int bcol = col0 + l15;
    const float bv = bias ? bias[bcol] : 0.0f;
#pragma unroll
    for (int i = 0; i < 8; ++i) {
        const int r = row0 + i + hl * 8;              // C layout: row = vgpr + 8*half
        const float vo = acc[i] + bv;
        if (!transposed) out[(size_t)r * ldOut + bcol] = (__bf16)vo;
        else             out[(size_t)bcol * ldOut + r] = (__bf16)vo;
    }
}

// ---------------------------------------------------------------------------
// Fused relation-attention: bias(emb) + QK^T + online softmax + PV.
// grid = N/16 blocks, block = 512 thr = 16 waves (wave g handles group g).
// ---------------------------------------------------------------------------
__global__ __launch_bounds__(512)
void attn_kernel(const __bf16* __restrict__ Q, const __bf16* __restrict__ K,
                 const __bf16* __restrict__ VT,
                 const float* __restrict__ rois_cur, const float* __restrict__ rois_ref,
                 const float* __restrict__ Wg_w, const float* __restrict__ Wg_b,
                 const float* __restrict__ Wv_b, float* __restrict__ out) {
    __shared__ float curg[16][4];            // w,h,cx,cy of current boxes
    __shared__ float refg[32][4];            // w,h,cx,cy of ref-box tile
    __shared__ float WgS[NGRP][DGRP];
    __shared__ float WgbS[NGRP];
    __shared__ float biasS[16][32][NGRP + 1];          // [n][m][g], padded
    __shared__ __align__(16) __bf16 pbuf[NGRP][16][32]; // per-wave P staging

    const int tid  = threadIdx.x;
    const int lane = tid & 31;
    const int g    = tid >> 5;
    const int l15  = lane & 15;
    const int hl   = lane >> 4;
    const int n0   = blockIdx.x * 16;

    for (int i = tid; i < NGRP * DGRP; i += 512) WgS[i >> 6][i & 63] = Wg_w[i];
    if (tid < NGRP) WgbS[tid] = Wg_b[tid];
    if (tid < 16) {
        const float x0 = rois_cur[(n0 + tid) * 4 + 0];
        const float y0 = rois_cur[(n0 + tid) * 4 + 1];
        const float x1 = rois_cur[(n0 + tid) * 4 + 2];
        const float y1 = rois_cur[(n0 + tid) * 4 + 3];
        curg[tid][0] = x1 - x0 + 1.0f;
        curg[tid][1] = y1 - y0 + 1.0f;
        curg[tid][2] = 0.5f * (x0 + x1);
        curg[tid][3] = 0.5f * (y0 + y1);
    }
    __syncthreads();

    // Preload Q A-fragments (16 rows x 64 dims = 2 fragments), fixed all loop.
    Frag qa[2];
    {
        const __bf16* qp = Q + (size_t)(n0 + l15) * FEATD + g * DGRP;
        qa[0].u[0] = *(const v4u*)(qp + hl * 8);
        qa[0].u[1] = *(const v4u*)(qp + 16 + hl * 8);
        qa[1].u[0] = *(const v4u*)(qp + 32 + hl * 8);
        qa[1].u[1] = *(const v4u*)(qp + 48 + hl * 8);
    }

    v8f o[4];
#pragma unroll
    for (int i = 0; i < 4; ++i) o[i] = (v8f){0.f,0.f,0.f,0.f,0.f,0.f,0.f,0.f};
    float Mr[8], Lr[8];
#pragma unroll
    for (int i = 0; i < 8; ++i) { Mr[i] = -1e30f; Lr[i] = 0.f; }

    // 1000^(-j/8) for j=0..7
    const float inv_dim[8] = {1.0f, 0.421696503f, 0.177827941f, 0.074989421f,
                              0.031622777f, 0.013335214f, 0.005623413f, 0.002371374f};

    for (int ms = 0; ms < M_REF; ms += 32) {
        if (tid < 32) {
            const float x0 = rois_ref[(ms + tid) * 4 + 0];
            const float y0 = rois_ref[(ms + tid) * 4 + 1];
            const float x1 = rois_ref[(ms + tid) * 4 + 2];
            const float y1 = rois_ref[(ms + tid) * 4 + 3];
            refg[tid][0] = x1 - x0 + 1.0f;
            refg[tid][1] = y1 - y0 + 1.0f;
            refg[tid][2] = 0.5f * (x0 + x1);
            refg[tid][3] = 0.5f * (y0 + y1);
        }
        __syncthreads();

        // --- position-embedding bias for all 16 groups: one (n,m) pair/thread
        {
            const int bn = tid >> 5, bm = tid & 31;
            const float w  = curg[bn][0], h  = curg[bn][1];
            const float cx = curg[bn][2], cy = curg[bn][3];
            const float wr = refg[bm][0], hr = refg[bm][1];
            const float cxr = refg[bm][2], cyr = refg[bm][3];
            float pos[4];
            pos[0] = __logf(fabsf((cx - cxr) / w) + 0.001f);
            pos[1] = __logf(fabsf((cy - cyr) / h) + 0.001f);
            pos[2] = __logf(w / wr);
            pos[3] = __logf(h / hr);
            float acc[NGRP];
#pragma unroll
            for (int gg = 0; gg < NGRP; ++gg) acc[gg] = WgbS[gg];
#pragma unroll
            for (int c = 0; c < 4; ++c) {
                const float p = pos[c] * 100.0f;
#pragma unroll
                for (int j = 0; j < 8; ++j) {
                    const float t  = p * inv_dim[j];
                    const float sv = __sinf(t);
                    const float cv = __cosf(t);
                    const int es = c * 16 + j, ec = c * 16 + 8 + j;
#pragma unroll
                    for (int gg = 0; gg < NGRP; ++gg)
                        acc[gg] += sv * WgS[gg][es] + cv * WgS[gg][ec];
                }
            }
#pragma unroll
            for (int gg = 0; gg < NGRP; ++gg)
                biasS[bn][bm][gg] = __logf(fmaxf(acc[gg], 0.0f) + 1e-6f);
        }
        __syncthreads();

        // --- S = Q * K^T for this group's 16x32 tile (4 WMMAs)
        v8f s[2];
#pragma unroll
        for (int mt = 0; mt < 2; ++mt) {
            v8f z = {0.f,0.f,0.f,0.f,0.f,0.f,0.f,0.f};
#pragma unroll
            for (int dc = 0; dc < 2; ++dc) {
                Frag kb;
                const __bf16* kp = K + (size_t)(ms + mt * 16 + l15) * FEATD
                                     + g * DGRP + dc * 32 + hl * 8;
                kb.u[0] = *(const v4u*)(kp);
                kb.u[1] = *(const v4u*)(kp + 16);
                z = wmma_bf16(qa[dc].v, kb.v, z);
            }
            s[mt] = z;
        }

        // --- online softmax (rows live in 16-lane halves; reduce with shfl_xor)
#pragma unroll
        for (int i = 0; i < 8; ++i) {
            const int r = i + hl * 8;
            float s0 = s[0][i] * 0.125f + biasS[r][l15][g];
            float s1 = s[1][i] * 0.125f + biasS[r][16 + l15][g];
            float v = fmaxf(s0, s1);
            v = fmaxf(v, __shfl_xor(v, 1, 32));
            v = fmaxf(v, __shfl_xor(v, 2, 32));
            v = fmaxf(v, __shfl_xor(v, 4, 32));
            v = fmaxf(v, __shfl_xor(v, 8, 32));
            const float mnew = fmaxf(Mr[i], v);
            const float sc   = __expf(Mr[i] - mnew);
            Mr[i] = mnew;
            const float p0 = __expf(s0 - mnew);
            const float p1 = __expf(s1 - mnew);
            float rs = p0 + p1;
            rs += __shfl_xor(rs, 1, 32);
            rs += __shfl_xor(rs, 2, 32);
            rs += __shfl_xor(rs, 4, 32);
            rs += __shfl_xor(rs, 8, 32);
            Lr[i] = Lr[i] * sc + rs;
            o[0][i] *= sc; o[1][i] *= sc; o[2][i] *= sc; o[3][i] *= sc;
            pbuf[g][r][l15]      = (__bf16)p0;
            pbuf[g][r][16 + l15] = (__bf16)p1;
        }
        __syncthreads();

        // --- O += P * V   (P re-laid-out via LDS into A fragment; VT rows are
        //     contiguous so the B fragment is two b128 loads per lane)
        Frag pa;
        pa.u[0] = *(const v4u*)&pbuf[g][l15][hl * 8];
        pa.u[1] = *(const v4u*)&pbuf[g][l15][16 + hl * 8];
#pragma unroll
        for (int dc = 0; dc < 4; ++dc) {
            Frag vb;
            const __bf16* vp = VT + (size_t)(g * DGRP + dc * 16 + l15) * M_REF
                                  + ms + hl * 8;
            vb.u[0] = *(const v4u*)(vp);
            vb.u[1] = *(const v4u*)(vp + 16);
            o[dc] = wmma_bf16(pa.v, vb.v, o[dc]);
        }
        if (ms + 32 < M_REF)
            __builtin_prefetch(K + (size_t)(ms + 32 + l15) * FEATD + g * DGRP, 0, 3);
    }

    // --- epilogue: normalize, add Wv bias, write f32 output (coalesced)
#pragma unroll
    for (int dc = 0; dc < 4; ++dc) {
#pragma unroll
        for (int i = 0; i < 8; ++i) {
            const int r = n0 + i + hl * 8;
            const int f = g * DGRP + dc * 16 + l15;
            out[(size_t)r * FEATD + f] = o[dc][i] / Lr[i] + Wv_b[f];
        }
    }
}

// ---------------------------------------------------------------------------
extern "C" void kernel_launch(void* const* d_in, const int* in_sizes, int n_in,
                              void* d_out, int out_size, void* d_ws, size_t ws_size,
                              hipStream_t stream) {
    const float* roi_feat = (const float*)d_in[0];
    const float* ref_feat = (const float*)d_in[1];
    const float* rois_cur = (const float*)d_in[2];
    const float* rois_ref = (const float*)d_in[3];
    const float* Wg_w     = (const float*)d_in[4];
    const float* Wg_b     = (const float*)d_in[5];
    const float* Wq_w     = (const float*)d_in[6];
    const float* Wq_b     = (const float*)d_in[7];
    const float* Wk_w     = (const float*)d_in[8];
    const float* Wk_b     = (const float*)d_in[9];
    const float* Wv_w     = (const float*)d_in[10];
    const float* Wv_b     = (const float*)d_in[11];
    float* out = (float*)d_out;

    char* w = (char*)d_ws;
    __bf16* qb  = (__bf16*)(w);                                   // 512*1024*2  = 1 MB
    __bf16* kb  = (__bf16*)(w + (size_t)1  * 1024 * 1024);        // 4096*1024*2 = 8 MB
    __bf16* vtb = (__bf16*)(w + (size_t)9  * 1024 * 1024);        // 1024*4096*2 = 8 MB

    // Q = roi_feat @ Wq^T + bq           -> [512, 1024] bf16
    gemm_nt_bf16<<<dim3(N_ROI / 32, FEATD / 64), 256, 0, stream>>>(
        roi_feat, Wq_w, Wq_b, qb, FEATD, 0);
    // K = ref_feat @ Wk^T + bk           -> [4096, 1024] bf16
    gemm_nt_bf16<<<dim3(M_REF / 32, FEATD / 64), 256, 0, stream>>>(
        ref_feat, Wk_w, Wk_b, kb, FEATD, 0);
    // V = ref_feat @ Wv^T (no bias), stored transposed -> [1024, 4096] bf16
    gemm_nt_bf16<<<dim3(M_REF / 32, FEATD / 64), 256, 0, stream>>>(
        ref_feat, Wv_w, nullptr, vtb, M_REF, 1);

    // Fused bias + softmax + PV
    attn_kernel<<<N_ROI / 16, 512, 0, stream>>>(
        qb, kb, vtb, rois_cur, rois_ref, Wg_w, Wg_b, Wv_b, out);
}
